// DEC_72086731096718
// MI455X (gfx1250) — compile-verified, matrix operand order checked by reading, not run on hardware
//
#include <hip/hip_runtime.h>

typedef __attribute__((ext_vector_type(16))) __bf16 v16bf;
typedef __attribute__((ext_vector_type(8)))  float  v8f;

#define D_DIM 512
#define K_DIM 2048
#define BM 16
#define NUM_DC (D_DIM / 32)   // 16 chunks of K=32
#define NT_WAVE 16            // 16 n-subtiles of 16 cols per wave (wave covers 256 cols)
#define TILE_BYTES (NUM_DC * 32 * 16 * 2)          // bytes per 16-col tile = 16KB
#define WAVE_BYTES ((size_t)NT_WAVE * TILE_BYTES)  // bytes per wave strip = 256KB

__device__ __forceinline__ unsigned short f2bf(float f) {
  unsigned int u = __float_as_uint(f);
  unsigned int r = u + 0x7FFFu + ((u >> 16) & 1u);   // round-to-nearest-even
  return (unsigned short)(r >> 16);
}

// Packed bf16 offset for element (row n within a 16-row tile t, depth d):
// block = t*NUM_DC + c ; lane = (n&15) | (hi<<4) ; elem = h*8+i
// d = c*32 + h*16 + hi*8 + i   -> matches WMMA 16-bit operand lane layout.
__device__ __forceinline__ size_t packed_off(int n, int d) {
  int t = n >> 4, c = d >> 5;
  int rem = d & 31, h = rem >> 4, rem2 = rem & 15;
  int hi = rem2 >> 3, i = rem2 & 7;
  int lane = (n & 15) | (hi << 4);
  return ((size_t)(t * NUM_DC + c) * 32 + lane) * 16 + h * 8 + i;
}

// ---- prep: centroids -> packed bf16 + c2 (one block of 256 per centroid row)
__global__ __launch_bounds__(256) void dec_prep_c(const float* __restrict__ cen,
                                                  unsigned short* __restrict__ cbf,
                                                  float* __restrict__ c2) {
  __shared__ float red[256];
  const int k = blockIdx.x, tid = threadIdx.x;
  float s = 0.f;
#pragma unroll
  for (int it = 0; it < 2; ++it) {
    int d = it * 256 + tid;
    float v = cen[(size_t)k * D_DIM + d];
    s += v * v;
    cbf[packed_off(k, d)] = f2bf(v);
  }
  red[tid] = s;
  __syncthreads();
  for (int off = 128; off > 0; off >>= 1) {
    if (tid < off) red[tid] += red[tid + off];
    __syncthreads();
  }
  if (tid == 0) c2[k] = red[0];
}

// ---- prep: x row sums of squares (one block per row)
__global__ __launch_bounds__(256) void dec_prep_x(const float* __restrict__ x,
                                                  float* __restrict__ x2) {
  __shared__ float red[256];
  const int b = blockIdx.x, tid = threadIdx.x;
  float s = 0.f;
#pragma unroll
  for (int it = 0; it < 2; ++it) {
    float v = x[(size_t)b * D_DIM + it * 256 + tid];
    s += v * v;
  }
  red[tid] = s;
  __syncthreads();
  for (int off = 128; off > 0; off >>= 1) {
    if (tid < off) red[tid] += red[tid + off];
    __syncthreads();
  }
  if (tid == 0) x2[b] = red[0];
}

// ---- fused GEMM + Student-t kernel + row normalization
// One workgroup = 16 rows x ALL 2048 columns. 8 waves, each wave: 256 columns.
__global__ __launch_bounds__(256) void dec_gemm(const float* __restrict__ x,
                                                const unsigned short* __restrict__ cbf,
                                                const float* __restrict__ c2,
                                                const float* __restrict__ x2,
                                                float* __restrict__ out) {
  __shared__ unsigned short xp[NUM_DC * 32 * 16];   // 16 KB packed bf16 A tile
  __shared__ float x2s[BM];
  __shared__ float part[256][8];                    // per-thread row partials
  __shared__ float rowinv[BM];

  const int tid = threadIdx.x;
  const int L = tid & 31;       // lane 0..31
  const int row0 = blockIdx.x * BM;

  // Stage x tile: 16x512 f32 -> packed bf16 in LDS
#pragma unroll
  for (int it = 0; it < 32; ++it) {
    int flat = it * 256 + tid;          // 0..8191
    int r = flat >> 9;                  // row in tile
    int d = flat & 511;
    float v = x[(size_t)(row0 + r) * D_DIM + d];
    xp[packed_off(r, d)] = f2bf(v);
  }
  if (tid < BM) x2s[tid] = x2[row0 + tid];
  __syncthreads();

  v8f acc[NT_WAVE];
#pragma unroll
  for (int j = 0; j < NT_WAVE; ++j) acc[j] = (v8f){0.f,0.f,0.f,0.f,0.f,0.f,0.f,0.f};

  // Wave-uniform base (SGPR) + single per-lane byte offset; all j-offsets are
  // compile-time immediates folded into global_load's 24-bit ioffset.
  const int wu = __builtin_amdgcn_readfirstlane(tid >> 5);   // wave id 0..7
  const char* bptr = (const char*)cbf + (size_t)wu * WAVE_BYTES + (size_t)L * 32;
  const char* aptr = (const char*)xp + (size_t)L * 32;

#pragma unroll 2
  for (int c = 0; c < NUM_DC; ++c) {
    v16bf a = *(const v16bf*)aptr;
#pragma unroll
    for (int j = 0; j < NT_WAVE; ++j) {
      v16bf bmat = *(const v16bf*)(bptr + (size_t)j * TILE_BYTES);
      acc[j] = __builtin_amdgcn_wmma_f32_16x16x32_bf16(
          false, a, false, bmat, (short)0, acc[j], false, false);
    }
    aptr += 1024;   // next K=32 chunk in LDS A tile
    bptr += 1024;   // next K=32 chunk in packed centroids
  }

  // Epilogue: d = x2 + c2 - 2*dot ; q = 1/(1+max(d,0)); accumulate row partials
  const int hioff = (L >> 4) << 3;      // 0 or 8: M offset for upper half-wave
  const int nlo = L & 15;
  float sume[8] = {0.f,0.f,0.f,0.f,0.f,0.f,0.f,0.f};
#pragma unroll
  for (int j = 0; j < NT_WAVE; ++j) {
    float c2v = c2[wu * 256 + j * 16 + nlo];
#pragma unroll
    for (int e = 0; e < 8; ++e) {
      float pre = x2s[hioff + e] + c2v;
      float dd = fmaxf(fmaf(-2.0f, acc[j][e], pre), 0.0f);
      float q = 1.0f / (1.0f + dd);
      acc[j][e] = q;
      sume[e] += q;
    }
  }
#pragma unroll
  for (int e = 0; e < 8; ++e) part[tid][e] = sume[e];
  __syncthreads();

  // Deterministic row-sum reduction: thread r (<16) owns row r
  if (tid < BM) {
    int e = tid & 7, half = tid >> 3;
    float s = 0.f;
    for (int ww = 0; ww < 8; ++ww)
#pragma unroll
      for (int l = 0; l < 16; ++l)
        s += part[ww * 32 + half * 16 + l][e];
    rowinv[tid] = 1.0f / s;
  }
  __syncthreads();

  // Scale and store (each b32 store covers two fully-used 64B segments)
#pragma unroll
  for (int e = 0; e < 8; ++e) {
    float inv = rowinv[hioff + e];
    size_t rbase = (size_t)(row0 + hioff + e) * K_DIM + wu * 256 + nlo;
#pragma unroll
    for (int j = 0; j < NT_WAVE; ++j) {
      out[rbase + j * 16] = acc[j][e] * inv;
    }
  }
}

extern "C" void kernel_launch(void* const* d_in, const int* in_sizes, int n_in,
                              void* d_out, int out_size, void* d_ws, size_t ws_size,
                              hipStream_t stream) {
  const float* x   = (const float*)d_in[0];
  const float* cen = (const float*)d_in[1];
  float* out = (float*)d_out;

  const int B = in_sizes[0] / D_DIM;   // 16384
  const int K = in_sizes[1] / D_DIM;   // 2048

  // workspace layout: [cbf packed bf16 K*D] [c2 K floats] [x2 B floats]
  unsigned short* cbf = (unsigned short*)d_ws;
  float* c2 = (float*)((char*)d_ws + (size_t)K * D_DIM * sizeof(unsigned short));
  float* x2 = c2 + K;

  dec_prep_c<<<K, 256, 0, stream>>>(cen, cbf, c2);
  dec_prep_x<<<B, 256, 0, stream>>>(x, x2);
  dec_gemm<<<B / BM, 256, 0, stream>>>(x, cbf, c2, x2, out);
}
